// Encoder_31241592111260
// MI455X (gfx1250) — matrix operand order repaired and checked by reading, not compiled, not run on hardware
//
#include <hip/hip_runtime.h>
#include <hip/hip_bf16.h>
#include <math.h>
#include <stdint.h>

// ---------------------------------------------------------------------------
// WaveNet encoder for MI455X (gfx1250, wave32, WMMA + TDM).
//   B=4, C_IN=1, T=49152, H=128, NCLS=256, 28 layers, K=2, OW=16386.
// Per-layer fused kernel: TDM async tensor-load stages both dilated taps into
// LDS (pad-enabled for alignment), BN-affine + transpose into the WMMA tile,
// gated conv = two K=256 v_wmma_f32_16x16x32_f16 GEMMs, tanh*sigmoid, 1x1
// WMMA GEMM, residual store + skip L2-atomics + BN batch-stat atomics.
// ---------------------------------------------------------------------------

#define Bc      4
#define Hc      128
#define Tc      49152
#define OWc     16386
#define NCLSc   256
#define NLAYERS 28
#define NTile   64          // timesteps per workgroup
#define LDP     264         // LDS row stride for 256-k act tiles (padded)
#define LDH     136         // LDS row stride for 128-k tiles (padded)
#define SSTR    72          // TDM staging row stride in f16 (64 data + 4DW pad)

#if __has_builtin(__builtin_amdgcn_tensor_load_to_lds)
#define HAVE_TDM 1
#endif

typedef __attribute__((ext_vector_type(16))) _Float16 v16h;
typedef __attribute__((ext_vector_type(8)))  _Float16 v8h;
typedef __attribute__((ext_vector_type(8)))  float    v8f;
typedef __attribute__((ext_vector_type(4)))  unsigned int u32x4;
typedef __attribute__((ext_vector_type(8)))  int i32x8;
typedef __attribute__((ext_vector_type(4)))  int i32x4;

__device__ __forceinline__ v16h pack16(v8h lo, v8h hi) {
  v16h r;
#pragma unroll
  for (int i = 0; i < 8; ++i) { r[i] = lo[i]; r[i + 8] = hi[i]; }
  return r;
}

// A fragment (16x32 f16, M x K): W row-major [M][ktot].
// lanes 0-15: row M=lane, K = kb+{0..15}; lanes 16-31: row M=lane-16,
// K = kb+{16..31}.
__device__ __forceinline__ v16h load_a(const _Float16* W, int ktot, int mBase,
                                       int kb, int lane) {
  const _Float16* p =
      W + (size_t)(mBase + (lane & 15)) * ktot + kb + ((lane & 16) ? 16 : 0);
  return pack16(*(const v8h*)p, *(const v8h*)(p + 8));
}

// B fragment (32x16 f16): caller passes &row[n][kb + (lane&16 ? 16 : 0)].
__device__ __forceinline__ v16h load_b(const _Float16* p) {
  return pack16(*(const v8h*)p, *(const v8h*)(p + 8));
}

__device__ __forceinline__ v8f wmma_f16(v16h a, v16h b, v8f c) {
  return __builtin_amdgcn_wmma_f32_16x16x32_f16(false, a, false, b, (short)0, c,
                                                false, false);
}

__device__ __forceinline__ float sigmoidf_(float x) {
  return 1.0f / (1.0f + __expf(-x));
}

#ifdef HAVE_TDM
// Issue one TDM 2D tile load: [rows=tile1][cols=tile0] f16, global row stride
// strideElems, into LDS at ldsByteAddr with 4-DWORD pad per 32-DWORD row
// (=> SSTR f16 per staged row). 6-arg builtin form (clang-23 toolchain).
__device__ __forceinline__ void tdm_load_2d(unsigned ldsByteAddr,
                                            unsigned long long gaBytes,
                                            unsigned tensorDim0,
                                            unsigned tile0, unsigned tile1,
                                            unsigned long long strideElems) {
  u32x4 g0;
  g0[0] = 1u;  // count=1, user descriptor, no gather
  g0[1] = ldsByteAddr;
  g0[2] = (unsigned)(gaBytes & 0xFFFFFFFFull);
  g0[3] = (unsigned)((gaBytes >> 32) & 0x01FFFFFFull) | (2u << 30);  // type=2
  i32x8 g1;
  // data_size=1 (2B) | pad_enable | pad_interval=4 (32 DW) | pad_amount=3 (4DW)
  g1[0] = (int)((1u << 16) | (1u << 20) | (4u << 22) | (3u << 25));
  g1[1] = (int)((tensorDim0 & 0xFFFFu) << 16);
  g1[2] = (int)(((tensorDim0 >> 16) & 0xFFFFu) | ((128u & 0xFFFFu) << 16));
  g1[3] = (int)((tile0 & 0xFFFFu) << 16);                 // dim1 hi=0, tile0
  g1[4] = (int)(tile1 & 0xFFFFu);                         // tile1, tile2=0
  g1[5] = (int)(strideElems & 0xFFFFFFFFull);
  g1[6] = (int)((strideElems >> 32) & 0xFFFFull);
  g1[7] = 0;
  i32x4 z4;
  z4[0] = 0; z4[1] = 0; z4[2] = 0; z4[3] = 0;
  i32x8 z8;
#pragma unroll
  for (int i = 0; i < 8; ++i) z8[i] = 0;
  __builtin_amdgcn_tensor_load_to_lds(g0, g1, z4, z4, z8, 0);
}
#endif

// ---------------------------------------------------------------------------
__global__ void k_init(float* a, size_t na, float* b, size_t nb) {
  size_t i = (size_t)blockIdx.x * blockDim.x + threadIdx.x;
  size_t stride = (size_t)gridDim.x * blockDim.x;
  for (size_t k = i; k < na; k += stride) a[k] = 0.0f;
  for (size_t k = i; k < nb; k += stride) b[k] = 0.0f;
}

// Convert stacked layer weights to f16; conv K-dim is [tap0 | tap1].
__global__ void k_convert_stack(const float* __restrict__ Wf,
                                const float* __restrict__ Wg,
                                const float* __restrict__ W1,
                                _Float16* __restrict__ wf16,
                                _Float16* __restrict__ wg16,
                                _Float16* __restrict__ w116) {
  const int l = blockIdx.x;  // 0..26
  for (int idx = threadIdx.x; idx < Hc * 2 * Hc; idx += blockDim.x) {
    int o = idx >> 8;
    int r = idx & 255;
    int tap = r >> 7;
    int i = r & 127;
    size_t src = ((size_t)(l * Hc + o) * Hc + i) * 2 + tap;
    wf16[(size_t)l * Hc * 256 + idx] = (_Float16)Wf[src];
    wg16[(size_t)l * Hc * 256 + idx] = (_Float16)Wg[src];
  }
  for (int idx = threadIdx.x; idx < Hc * Hc; idx += blockDim.x)
    w116[(size_t)l * Hc * Hc + idx] = (_Float16)W1[(size_t)l * Hc * Hc + idx];
}

__global__ void k_convert_head(const float* __restrict__ W10,
                               const float* __restrict__ W11,
                               const float* __restrict__ W12,
                               const float* __restrict__ Whc,
                               _Float16* __restrict__ w010,
                               _Float16* __restrict__ w11h,
                               _Float16* __restrict__ w12h,
                               _Float16* __restrict__ whch) {
  for (int idx = threadIdx.x; idx < Hc * Hc; idx += blockDim.x) {
    w010[idx] = (_Float16)W10[idx];
    w11h[idx] = (_Float16)W11[idx];
    w12h[idx] = (_Float16)W12[idx];
  }
  for (int idx = threadIdx.x; idx < NCLSc * 2 * Hc; idx += blockDim.x) {
    int o = idx >> 8;
    int r = idx & 255;
    int tap = r >> 7;
    int c = r & 127;
    whch[idx] = (_Float16)Whc[((size_t)(o * Hc + c)) * 2 + tap];
  }
}

// ---------------------------------------------------------------------------
// Layer 0: in_channels = 1, gating is cheap VALU; 1x1 conv uses WMMA.
__global__ __launch_bounds__(256) void k_layer0(
    const float* __restrict__ x, const float* __restrict__ Wf0,
    const float* __restrict__ bf0, const float* __restrict__ Wg0,
    const float* __restrict__ bg0, const _Float16* __restrict__ W1,
    const float* __restrict__ b1, _Float16* __restrict__ actOut,
    float* __restrict__ skipsum, float* __restrict__ stats) {
  __shared__ _Float16 sH[NTile * LDH];
  __shared__ float sX[NTile + 1];
  __shared__ float sStat[256];
  const int tid = threadIdx.x;
  const int lane = tid & 31;
  const int wave = tid >> 5;
  const int tilesPerB = Tc / NTile;
  const int bIdx = blockIdx.x / tilesPerB;
  const int t0 = (blockIdx.x % tilesPerB) * NTile;

  sStat[tid] = 0.0f;
  if (tid <= NTile) {
    int t = t0 + tid - 1;
    sX[tid] = (t >= 0) ? x[(size_t)bIdx * Tc + t] : 0.0f;
  }
  __syncthreads();

  for (int idx = tid; idx < Hc * NTile; idx += 256) {
    int c = idx >> 6;
    int n = idx & (NTile - 1);
    float xp = sX[n], xc = sX[n + 1];
    float f = tanhf(Wf0[c * 2] * xp + Wf0[c * 2 + 1] * xc + bf0[c]);
    float g = sigmoidf_(Wg0[c * 2] * xp + Wg0[c * 2 + 1] * xc + bg0[c]);
    sH[n * LDH + c] = (_Float16)(f * g);
  }
  __syncthreads();

  const int mBase = wave * 16;
  const int mOff = (lane >> 4) * 8;
  const int nl = lane & 15;
  const int kSel = (lane & 16) ? 16 : 0;
  const size_t skipBase = (size_t)bIdx * Hc * OWc;
  __builtin_prefetch(W1 + (size_t)mBase * Hc, 0, 1);

  v8f acc[4] = {};
#pragma unroll
  for (int ks = 0; ks < 4; ++ks) {
    int kb = ks * 32;
    v16h a = load_a(W1, Hc, mBase, kb, lane);
#pragma unroll
    for (int nt = 0; nt < 4; ++nt) {
      v16h bb = load_b(sH + (nt * 16 + nl) * LDH + kb + kSel);
      acc[nt] = wmma_f16(a, bb, acc[nt]);
    }
  }
#pragma unroll
  for (int nt = 0; nt < 4; ++nt) {
    int n = nt * 16 + nl;
    int t = t0 + n;
    float xv = sX[n + 1];
#pragma unroll
    for (int j = 0; j < 8; ++j) {
      int m = mBase + mOff + j;
      float skip = acc[nt][j] + b1[m];
      float r = skip + xv;  // residual (x broadcast over channels)
      actOut[((size_t)bIdx * Hc + m) * Tc + t] = (_Float16)r;
      atomicAdd(&sStat[m], r);
      atomicAdd(&sStat[128 + m], r * r);
      if (t >= Tc - OWc)
        atomicAdd(&skipsum[skipBase + (size_t)m * OWc + (t - (Tc - OWc))], skip);
    }
  }
  __syncthreads();
  atomicAdd(&stats[tid], sStat[tid]);
}

// ---------------------------------------------------------------------------
// Generic fused layer (layers 1..27). BN of previous layer applied on load.
__global__ __launch_bounds__(256) void k_layer(
    const _Float16* __restrict__ actIn, const float* __restrict__ bn,
    const float* __restrict__ gamma, const float* __restrict__ beta,
    const _Float16* __restrict__ Wf, const _Float16* __restrict__ Wg,
    const _Float16* __restrict__ W1, const float* __restrict__ bf,
    const float* __restrict__ bg, const float* __restrict__ b1, int dil,
    _Float16* __restrict__ actOut, float* __restrict__ skipsum,
    float* __restrict__ stats) {
  __shared__ _Float16 sAct[NTile * LDP];  // [n][k]: k<128 tap(t-d), k>=128 tap(t)
  __shared__ _Float16 sH[NTile * LDH];
  __shared__ float sStat[256];
#ifdef HAVE_TDM
  __shared__ _Float16 sStage[2 * Hc * SSTR];  // TDM staging, [tap][c][SSTR]
#endif
  const int tid = threadIdx.x;
  const int lane = tid & 31;
  const int wave = tid >> 5;
  const int tilesPerB = Tc / NTile;
  const int bIdx = blockIdx.x / tilesPerB;
  const int t0 = (blockIdx.x % tilesPerB) * NTile;

  sStat[tid] = 0.0f;

#ifdef HAVE_TDM
  // Stage 0: TDM DMA both dilated taps (raw f16) into padded LDS staging.
  if (tid < 32) {
#pragma unroll
    for (int tap = 0; tap < 2; ++tap) {
      int back = tap ? 0 : dil;
      int cs = t0 - back;
      if (cs < 0) cs = 0;  // clamp; shifted columns zero-filled in transpose
      unsigned lds = (unsigned)(uintptr_t)(sStage + (size_t)tap * Hc * SSTR);
      unsigned long long ga =
          (unsigned long long)(uintptr_t)(actIn + (size_t)bIdx * Hc * Tc + cs);
      tdm_load_2d(lds, ga, (unsigned)(Tc - cs), NTile, Hc,
                  (unsigned long long)Tc);
    }
    __builtin_amdgcn_s_wait_tensorcnt(0);
  }
  __syncthreads();
#endif

  // Stage 1: BN-affine + transpose into the [n][k] WMMA tile.
  {
    int c = tid & 127;
    int tap = tid >> 7;  // 0: t-dil tap, 1: current tap
    int back = tap ? 0 : dil;
    int shift = (t0 - back < 0) ? (back - t0) : 0;
    if (shift > NTile) shift = NTile;
    float sc = bn[128 + c] * gamma[c];  // rstd * gamma
    float sh = beta[c] - bn[c] * sc;    // beta - mean*rstd*gamma
    _Float16* dst = sAct + tap * Hc + c;
    for (int n = 0; n < shift; ++n) dst[n * LDP] = (_Float16)0.0f;  // causal pad
#ifdef HAVE_TDM
    const _Float16* stg = sStage + ((size_t)tap * Hc + c) * SSTR;
#pragma unroll
    for (int ch = 0; ch < 8; ++ch) {
      v8h v = *(const v8h*)(stg + ch * 8);  // 16B-aligned staged row chunk
#pragma unroll
      for (int j = 0; j < 8; ++j) {
        int n = ch * 8 + j + shift;
        if (n < NTile) dst[n * LDP] = (_Float16)((float)v[j] * sc + sh);
      }
    }
#else
    int cs = t0 - back;
    if (cs < 0) cs = 0;
    const _Float16* src = actIn + ((size_t)bIdx * Hc + c) * Tc + cs;
    for (int n = shift; n < NTile; ++n)
      dst[n * LDP] = (_Float16)((float)src[n - shift] * sc + sh);
#endif
  }
  __syncthreads();

  const int mBase = wave * 16;
  const int mOff = (lane >> 4) * 8;
  const int nl = lane & 15;
  const int kSel = (lane & 16) ? 16 : 0;
  __builtin_prefetch(Wf + (size_t)mBase * 256, 0, 1);
  __builtin_prefetch(Wg + (size_t)mBase * 256, 0, 1);

  // Stage 2: gated dilated conv: M=16,N=64,K=256 per wave; K-outer so each
  // A-fragment is fetched once and feeds 4 WMMAs.
  {
    v8f accF[4] = {};
    v8f accG[4] = {};
#pragma unroll
    for (int ks = 0; ks < 8; ++ks) {
      int kb = ks * 32;
      v16h af = load_a(Wf, 256, mBase, kb, lane);
      v16h ag = load_a(Wg, 256, mBase, kb, lane);
#pragma unroll
      for (int nt = 0; nt < 4; ++nt) {
        v16h bb = load_b(sAct + (nt * 16 + nl) * LDP + kb + kSel);
        accF[nt] = wmma_f16(af, bb, accF[nt]);
        accG[nt] = wmma_f16(ag, bb, accG[nt]);
      }
    }
#pragma unroll
    for (int nt = 0; nt < 4; ++nt) {
      int n = nt * 16 + nl;
#pragma unroll
      for (int j = 0; j < 8; ++j) {
        int m = mBase + mOff + j;
        float f = tanhf(accF[nt][j] + bf[m]);
        float g = sigmoidf_(accG[nt][j] + bg[m]);
        sH[n * LDH + m] = (_Float16)(f * g);
      }
    }
  }
  __syncthreads();

  // Stage 3: 1x1 conv (K=128 WMMA) + residual + skip atomics + BN stats.
  {
    const size_t skipBase = (size_t)bIdx * Hc * OWc;
    v8f acc[4] = {};
#pragma unroll
    for (int ks = 0; ks < 4; ++ks) {
      int kb = ks * 32;
      v16h a = load_a(W1, Hc, mBase, kb, lane);
#pragma unroll
      for (int nt = 0; nt < 4; ++nt) {
        v16h bb = load_b(sH + (nt * 16 + nl) * LDH + kb + kSel);
        acc[nt] = wmma_f16(a, bb, acc[nt]);
      }
    }
#pragma unroll
    for (int nt = 0; nt < 4; ++nt) {
      int n = nt * 16 + nl;
      int t = t0 + n;
#pragma unroll
      for (int j = 0; j < 8; ++j) {
        int m = mBase + mOff + j;
        float skip = acc[nt][j] + b1[m];
        float xv = (float)sAct[n * LDP + Hc + m];  // BN'd current-tap input
        float r = skip + xv;
        actOut[((size_t)bIdx * Hc + m) * Tc + t] = (_Float16)r;
        atomicAdd(&sStat[m], r);
        atomicAdd(&sStat[128 + m], r * r);
        if (t >= Tc - OWc)
          atomicAdd(&skipsum[skipBase + (size_t)m * OWc + (t - (Tc - OWc))],
                    skip);
      }
    }
  }
  __syncthreads();
  atomicAdd(&stats[tid], sStat[tid]);
}

// ---------------------------------------------------------------------------
__global__ void k_bn_finalize(const float* __restrict__ st,
                              float* __restrict__ bn) {
  int c = threadIdx.x;
  if (c < Hc) {
    const float inv = 1.0f / ((float)Bc * (float)Tc);
    float mean = st[c] * inv;
    float var = st[128 + c] * inv - mean * mean;  // biased, matches reference
    bn[c] = mean;
    bn[128 + c] = rsqrtf(var + 1e-5f);
  }
}

// ---------------------------------------------------------------------------
// Head 1x1 ReLU GEMM: out = relu(W @ in + b). in is f32 (skip_sum) or f16.
__global__ __launch_bounds__(256) void k_head_gemm(
    const float* __restrict__ inF32, const _Float16* __restrict__ inF16,
    const _Float16* __restrict__ W, const float* __restrict__ bias,
    _Float16* __restrict__ outp) {
  __shared__ _Float16 sIn[NTile * LDH];
  const int tid = threadIdx.x;
  const int lane = tid & 31;
  const int wave = tid >> 5;
  const int tilesPerB = (OWc + NTile - 1) / NTile;
  const int bIdx = blockIdx.x / tilesPerB;
  const int t0 = (blockIdx.x % tilesPerB) * NTile;

  for (int idx = tid; idx < NTile * Hc; idx += 256) {
    int n = idx & (NTile - 1);
    int c = idx >> 6;
    int t = t0 + n;
    float v = 0.0f;
    if (t < OWc) {
      size_t src = ((size_t)bIdx * Hc + c) * OWc + t;
      v = inF32 ? inF32[src] : (float)inF16[src];
    }
    sIn[n * LDH + c] = (_Float16)v;
  }
  __syncthreads();

  const int mBase = wave * 16;
  const int mOff = (lane >> 4) * 8;
  const int nl = lane & 15;
  const int kSel = (lane & 16) ? 16 : 0;
  v8f acc[4] = {};
#pragma unroll
  for (int ks = 0; ks < 4; ++ks) {
    int kb = ks * 32;
    v16h a = load_a(W, Hc, mBase, kb, lane);
#pragma unroll
    for (int nt = 0; nt < 4; ++nt) {
      v16h bb = load_b(sIn + (nt * 16 + nl) * LDH + kb + kSel);
      acc[nt] = wmma_f16(a, bb, acc[nt]);
    }
  }
#pragma unroll
  for (int nt = 0; nt < 4; ++nt) {
    int n = nt * 16 + nl;
    int t = t0 + n;
    if (t < OWc) {
#pragma unroll
      for (int j = 0; j < 8; ++j) {
        int m = mBase + mOff + j;
        outp[((size_t)bIdx * Hc + m) * OWc + t] =
            (_Float16)fmaxf(acc[nt][j] + bias[m], 0.0f);
      }
    }
  }
}

// Final K=2 conv to 256 classes: taps t and t+1 stacked in K (K=256), M=256.
__global__ __launch_bounds__(256) void k_head_conv(
    const _Float16* __restrict__ y2, const _Float16* __restrict__ W,
    const float* __restrict__ bias, float* __restrict__ outp) {
  __shared__ _Float16 sIn[NTile * LDP];
  const int tid = threadIdx.x;
  const int lane = tid & 31;
  const int wave = tid >> 5;
  const int OutW = OWc - 1;
  const int tilesPerB = (OutW + NTile - 1) / NTile;
  const int bIdx = blockIdx.x / tilesPerB;
  const int t0 = (blockIdx.x % tilesPerB) * NTile;

  for (int idx = tid; idx < NTile * 256; idx += 256) {
    int n = idx & (NTile - 1);
    int k = idx >> 6;
    int c = k & 127;
    int tap = k >> 7;
    int t = t0 + n + tap;
    float v = (t < OWc) ? (float)y2[((size_t)bIdx * Hc + c) * OWc + t] : 0.0f;
    sIn[n * LDP + k] = (_Float16)v;
  }
  __syncthreads();

  const int mOff = (lane >> 4) * 8;
  const int nl = lane & 15;
  const int kSel = (lane & 16) ? 16 : 0;
  for (int mi = 0; mi < 2; ++mi) {
    int mBase = (wave * 2 + mi) * 16;  // M = 256 -> 16 tiles over 8 waves
    v8f acc[4] = {};
#pragma unroll
    for (int ks = 0; ks < 8; ++ks) {
      int kb = ks * 32;
      v16h a = load_a(W, 256, mBase, kb, lane);
#pragma unroll
      for (int nt = 0; nt < 4; ++nt) {
        v16h bb = load_b(sIn + (nt * 16 + nl) * LDP + kb + kSel);
        acc[nt] = wmma_f16(a, bb, acc[nt]);
      }
    }
#pragma unroll
    for (int nt = 0; nt < 4; ++nt) {
      int n = nt * 16 + nl;
      int t = t0 + n;
      if (t < OutW) {
#pragma unroll
        for (int j = 0; j < 8; ++j) {
          int m = mBase + mOff + j;
          outp[((size_t)bIdx * NCLSc + m) * OutW + t] = acc[nt][j] + bias[m];
        }
      }
    }
  }
}

// ---------------------------------------------------------------------------
extern "C" void kernel_launch(void* const* d_in, const int* in_sizes, int n_in,
                              void* d_out, int out_size, void* d_ws,
                              size_t ws_size, hipStream_t stream) {
  (void)in_sizes; (void)n_in; (void)out_size; (void)ws_size;
  const float* x    = (const float*)d_in[0];
  const float* Wf0  = (const float*)d_in[1];
  const float* bf0  = (const float*)d_in[2];
  const float* Wg0  = (const float*)d_in[3];
  const float* bg0  = (const float*)d_in[4];
  const float* W10  = (const float*)d_in[5];
  const float* b10  = (const float*)d_in[6];
  const float* Wf   = (const float*)d_in[7];
  const float* bf   = (const float*)d_in[8];
  const float* Wg   = (const float*)d_in[9];
  const float* bg   = (const float*)d_in[10];
  const float* W1   = (const float*)d_in[11];
  const float* b1   = (const float*)d_in[12];
  const float* gamma= (const float*)d_in[13];
  const float* beta = (const float*)d_in[14];
  const float* W11  = (const float*)d_in[15];
  const float* b11  = (const float*)d_in[16];
  const float* W12  = (const float*)d_in[17];
  const float* b12  = (const float*)d_in[18];
  const float* Whc  = (const float*)d_in[19];
  const float* bhc  = (const float*)d_in[20];
  float* out = (float*)d_out;

  // Carve workspace (~140 MB): f16 weights, skip_sum, stats, bn, act ping-pong.
  char* ws = (char*)d_ws;
  size_t off = 0;
  auto carve = [&](size_t bytes) -> char* {
    off = (off + 255) & ~(size_t)255;
    char* p = ws + off;
    off += bytes;
    return p;
  };
  _Float16* wf16 = (_Float16*)carve((size_t)27 * Hc * 256 * 2);
  _Float16* wg16 = (_Float16*)carve((size_t)27 * Hc * 256 * 2);
  _Float16* w116 = (_Float16*)carve((size_t)27 * Hc * Hc * 2);
  _Float16* w010 = (_Float16*)carve((size_t)Hc * Hc * 2);
  _Float16* w11h = (_Float16*)carve((size_t)Hc * Hc * 2);
  _Float16* w12h = (_Float16*)carve((size_t)Hc * Hc * 2);
  _Float16* whch = (_Float16*)carve((size_t)NCLSc * 256 * 2);
  float* skipsum = (float*)carve((size_t)Bc * Hc * OWc * 4);
  float* stats   = (float*)carve((size_t)NLAYERS * 256 * 4);
  float* bnbuf   = (float*)carve((size_t)NLAYERS * 256 * 4);
  _Float16* actA = (_Float16*)carve((size_t)Bc * Hc * Tc * 2);
  _Float16* actB = (_Float16*)carve((size_t)Bc * Hc * Tc * 2);

  k_init<<<dim3(512), dim3(256), 0, stream>>>(skipsum, (size_t)Bc * Hc * OWc,
                                              stats, (size_t)NLAYERS * 256);
  k_convert_stack<<<dim3(27), dim3(256), 0, stream>>>(Wf, Wg, W1, wf16, wg16,
                                                      w116);
  k_convert_head<<<dim3(1), dim3(256), 0, stream>>>(W10, W11, W12, Whc, w010,
                                                    w11h, w12h, whch);

  const int layerBlocks = Bc * (Tc / NTile);  // 3072
  k_layer0<<<dim3(layerBlocks), dim3(256), 0, stream>>>(
      x, Wf0, bf0, Wg0, bg0, w010, b10, actA, skipsum, stats);
  k_bn_finalize<<<dim3(1), dim3(128), 0, stream>>>(stats, bnbuf);

  const _Float16* cur = actA;
  _Float16* nxt = actB;
  for (int i = 1; i < NLAYERS; ++i) {
    int l = i - 1;                 // stacked-weight index
    int dil = 1 << (i % 14);       // DILATIONS[i]
    k_layer<<<dim3(layerBlocks), dim3(256), 0, stream>>>(
        cur, bnbuf + (size_t)l * 256, gamma + (size_t)l * Hc,
        beta + (size_t)l * Hc, wf16 + (size_t)l * Hc * 256,
        wg16 + (size_t)l * Hc * 256, w116 + (size_t)l * Hc * Hc,
        bf + (size_t)l * Hc, bg + (size_t)l * Hc, b1 + (size_t)l * Hc, dil,
        nxt, skipsum, stats + (size_t)i * 256);
    if (i < NLAYERS - 1)  // last layer's BN output is never consumed
      k_bn_finalize<<<dim3(1), dim3(128), 0, stream>>>(
          stats + (size_t)i * 256, bnbuf + (size_t)i * 256);
    _Float16* tmp = (_Float16*)cur;
    cur = nxt;
    nxt = tmp;
  }

  // Head: nxt (old ping buffer) is free for y1; cur (last residual) for y2.
  const int headBlocks = Bc * ((OWc + NTile - 1) / NTile);  // 4 * 257
  _Float16* y1 = nxt;
  _Float16* y2 = (_Float16*)cur;
  k_head_gemm<<<dim3(headBlocks), dim3(256), 0, stream>>>(
      skipsum, (const _Float16*)nullptr, w11h, b11, y1);
  k_head_gemm<<<dim3(headBlocks), dim3(256), 0, stream>>>(
      (const float*)nullptr, y1, w12h, b12, y2);
  k_head_conv<<<dim3(headBlocks), dim3(256), 0, stream>>>(y2, whch, bhc, out);
}
